// NonLocalBlock2D_44066364457115
// MI455X (gfx1250) — matrix operand order repaired and checked by reading, not compiled, not run on hardware
//
#include <hip/hip_runtime.h>

// ---------------------------------------------------------------------------
// NonLocalBlock2D for MI455X (gfx1250): all GEMMs via v_wmma_f32_16x16x32_bf16,
// fragment-order LDS staging (ds_load_b128 fragment reads), dword-packed
// bf16 conversion/transpose via v_perm_b32.
// B=16, H=W=128, C=128, f=64, N=H*W=16384
// ---------------------------------------------------------------------------

#define BATCH  16
#define NPIX   16384   // H*W
#define CIN    128
#define FDIM   64
#define KSPLIT 32      // split-K factor for the Gram stage
#define BN_EPS 1e-3f

typedef __attribute__((ext_vector_type(16))) __bf16          v16bf;
typedef __attribute__((ext_vector_type(16))) unsigned short  v16u;
typedef __attribute__((ext_vector_type(8)))  float           v8f;

// scalar float -> bf16 bits, round-to-nearest-even (epilogues only)
static __device__ __forceinline__ unsigned short f2bf(float f) {
  unsigned int u = __builtin_bit_cast(unsigned int, f);
  u += 0x7fffu + ((u >> 16) & 1u);
  return (unsigned short)(u >> 16);
}

// two floats -> packed 2xbf16 dword (RNE), merged with one v_perm_b32
static __device__ __forceinline__ unsigned int f2bf2(float lo, float hi) {
  unsigned int ul = __builtin_bit_cast(unsigned int, lo);
  unsigned int uh = __builtin_bit_cast(unsigned int, hi);
  ul += 0x7fffu + ((ul >> 16) & 1u);
  uh += 0x7fffu + ((uh >> 16) & 1u);
  // result = { uh[31:16], ul[31:16] }
  return __builtin_amdgcn_perm(uh, ul, 0x07060302u);
}

// 4x4 u16 transpose on packed dwords: rows a..d (each uint2 = 4 bf16),
// columns out q0..q3. 8x v_perm_b32.
static __device__ __forceinline__ void tr4x4(uint2 a, uint2 b, uint2 c, uint2 d,
                                             uint2& q0, uint2& q1, uint2& q2, uint2& q3) {
  q0.x = __builtin_amdgcn_perm(b.x, a.x, 0x05040100u);  // {b.e0, a.e0}
  q0.y = __builtin_amdgcn_perm(d.x, c.x, 0x05040100u);  // {d.e0, c.e0}
  q1.x = __builtin_amdgcn_perm(b.x, a.x, 0x07060302u);  // {b.e1, a.e1}
  q1.y = __builtin_amdgcn_perm(d.x, c.x, 0x07060302u);
  q2.x = __builtin_amdgcn_perm(b.y, a.y, 0x05040100u);  // {b.e2, a.e2}
  q2.y = __builtin_amdgcn_perm(d.y, c.y, 0x05040100u);
  q3.x = __builtin_amdgcn_perm(b.y, a.y, 0x07060302u);  // {b.e3, a.e3}
  q3.y = __builtin_amdgcn_perm(d.y, c.y, 0x07060302u);
}

static __device__ __forceinline__ v8f wmma_bf16(v16bf a, v16bf b, v8f c) {
  return __builtin_amdgcn_wmma_f32_16x16x32_bf16(
      false, a, false, b, (short)0, c, false, false);
}

// Fragment read: lane's 16 bf16 are contiguous 32 B -> 2x ds_load_b128.
static __device__ __forceinline__ v16bf ldfrag(const unsigned short* p) {
  return __builtin_bit_cast(v16bf, *(const v16u*)p);
}

// Fragment-order index maps (wave32 WMMA 16x16x32 bf16, contiguous k order):
//  A (16x32, MxK):  lane = (k>>4)*16 + m,             elem = k&15
//  B (32x16, KxN):  lane = (k>>4)*16 + n,             elem = k&15
//  C (16x16 f32):   row m = r + 8*(lane>>4), col n = lane&15
// For k aligned to 4 (k&7 in {0,4}), 4 consecutive k are contiguous elems.

// ---------------------------------------------------------------------------
// Kernel 1: projection GEMM  out_bf16[b,n,f] = X[b,n,:] @ W[:,f] + bias
// grid(N/128, B), block 256 (8 waves). Wave: 16 rows x 64 cols.
// ---------------------------------------------------------------------------
__global__ __launch_bounds__(256) void k_proj(const float* __restrict__ X,
                                              const float* __restrict__ W,
                                              const float* __restrict__ bias,
                                              unsigned short* __restrict__ out) {
  const int b    = blockIdx.y;
  const int row0 = blockIdx.x * 128;
  const int tid  = threadIdx.x;
  const int lane = tid & 31, wave = tid >> 5;

  __shared__ __align__(32) unsigned short ldsA[8][32][16];  // X tile, frag order
  __shared__ __align__(32) unsigned short ldsB[4][32][16];  // W tile, frag order

  v8f acc[4];
#pragma unroll
  for (int j = 0; j < 4; ++j)
#pragma unroll
    for (int r = 0; r < 8; ++r) acc[j][r] = 0.0f;

  for (int kk = 0; kk < CIN; kk += 32) {
    // stage X tile (128 rows x 32 k), fp32 -> packed bf16, direct frag order
#pragma unroll
    for (int it = 0; it < 4; ++it) {
      const int lin = tid * 4 + it * 1024;
      const int r = lin >> 5, c = lin & 31;
      const float4 v = *(const float4*)(X + ((size_t)b * NPIX + row0 + r) * CIN + kk + c);
      uint2 q;
      q.x = f2bf2(v.x, v.y);
      q.y = f2bf2(v.z, v.w);
      *(uint2*)&ldsA[r >> 4][(c >> 4) * 16 + (r & 15)][c & 15] = q;
    }
    // stage W tile (32 k x 64 n): pack rows, 4x4 perm transpose, threads 0..127
    if (tid < 128) {
      const int k0 = (tid >> 4) * 4, c0 = (tid & 15) * 4;
      const float* ws = W + (size_t)(kk + k0) * FDIM + c0;
      const float4 w0 = *(const float4*)(ws);
      const float4 w1 = *(const float4*)(ws + FDIM);
      const float4 w2 = *(const float4*)(ws + 2 * FDIM);
      const float4 w3 = *(const float4*)(ws + 3 * FDIM);
      const uint2 a = {f2bf2(w0.x, w0.y), f2bf2(w0.z, w0.w)};
      const uint2 bb = {f2bf2(w1.x, w1.y), f2bf2(w1.z, w1.w)};
      const uint2 cc = {f2bf2(w2.x, w2.y), f2bf2(w2.z, w2.w)};
      const uint2 dd = {f2bf2(w3.x, w3.y), f2bf2(w3.z, w3.w)};
      uint2 q0, q1, q2, q3;
      tr4x4(a, bb, cc, dd, q0, q1, q2, q3);
      const int jt = c0 >> 4, l0 = (k0 >> 4) * 16 + (c0 & 15), e = k0 & 15;
      *(uint2*)&ldsB[jt][l0 + 0][e] = q0;
      *(uint2*)&ldsB[jt][l0 + 1][e] = q1;
      *(uint2*)&ldsB[jt][l0 + 2][e] = q2;
      *(uint2*)&ldsB[jt][l0 + 3][e] = q3;
    }
    if (kk + 32 < CIN)
      __builtin_prefetch(X + ((size_t)b * NPIX + row0 + (tid >> 1)) * CIN + kk + 32, 0, 1);
    __syncthreads();

    const v16bf a = ldfrag(&ldsA[wave][lane][0]);
#pragma unroll
    for (int jt = 0; jt < 4; ++jt)
      acc[jt] = wmma_bf16(a, ldfrag(&ldsB[jt][lane][0]), acc[jt]);
    __syncthreads();
  }

  const int hf = lane >> 4, nc = lane & 15;
#pragma unroll
  for (int jt = 0; jt < 4; ++jt) {
    const int col = jt * 16 + nc;
    const float bv = bias[col];
#pragma unroll
    for (int r = 0; r < 8; ++r) {
      const int m = r + 8 * hf;
      out[((size_t)b * NPIX + row0 + wave * 16 + m) * FDIM + col] = f2bf(acc[jt][r] + bv);
    }
  }
}

// ---------------------------------------------------------------------------
// Kernel 2: split-K Gram  fm_part[b,s,i,j] = sum_{n in chunk s} th[n,i]*ph[n,j]
// grid(KSPLIT, B), block 256. 16 output 16x16 tiles, 2 per wave.
// Threads 0..127 stage theta (A, transposed), 128..255 stage phi (B).
// ---------------------------------------------------------------------------
__global__ __launch_bounds__(256) void k_gram(const unsigned short* __restrict__ tx,
                                              const unsigned short* __restrict__ ty,
                                              float* __restrict__ fmp) {
  const int b   = blockIdx.y;
  const int sk  = blockIdx.x;
  const int tid = threadIdx.x;
  const int lane = tid & 31, wave = tid >> 5;

  __shared__ __align__(32) unsigned short ldsAt[4][32][16];  // theta^T frag order
  __shared__ __align__(32) unsigned short ldsB[4][32][16];   // phi frag order

  const unsigned short* txb = tx + (size_t)b * NPIX * FDIM;
  const unsigned short* tyb = ty + (size_t)b * NPIX * FDIM;

  const int t0 = wave * 2;
  int i0[2], j0[2];
#pragma unroll
  for (int s = 0; s < 2; ++s) {
    i0[s] = ((t0 + s) >> 2) * 16;
    j0[s] = ((t0 + s) & 3) * 16;
  }
  v8f acc[2];
#pragma unroll
  for (int s = 0; s < 2; ++s)
#pragma unroll
    for (int r = 0; r < 8; ++r) acc[s][r] = 0.0f;

  const int sel = tid >> 7, blk = tid & 127;
  const int k0 = (blk >> 4) * 4, c0 = (blk & 15) * 4;
  const unsigned short* srcb = sel ? tyb : txb;

  const int nend = sk * (NPIX / KSPLIT) + (NPIX / KSPLIT);
  for (int n0 = sk * (NPIX / KSPLIT); n0 < nend; n0 += 32) {
    // 4x4 perm-transpose staging (one block per thread)
    const unsigned short* src = srcb + (size_t)(n0 + k0) * FDIM + c0;
    const uint2 r0 = *(const uint2*)(src);
    const uint2 r1 = *(const uint2*)(src + FDIM);
    const uint2 r2 = *(const uint2*)(src + 2 * FDIM);
    const uint2 r3 = *(const uint2*)(src + 3 * FDIM);
    uint2 q0, q1, q2, q3;
    tr4x4(r0, r1, r2, r3, q0, q1, q2, q3);
    if (sel == 0) {  // A^T: lane = (k>>4)*16 + i_local, elem = k&15
      const int it = c0 >> 4, l0 = ((k0 >> 4) * 16) + (c0 & 15);
      const int e = k0 & 15;
      *(uint2*)&ldsAt[it][l0 + 0][e] = q0;
      *(uint2*)&ldsAt[it][l0 + 1][e] = q1;
      *(uint2*)&ldsAt[it][l0 + 2][e] = q2;
      *(uint2*)&ldsAt[it][l0 + 3][e] = q3;
    } else {         // B: lane = (k>>4)*16 + j_local, elem = k&15
      const int jt = c0 >> 4, l0 = (k0 >> 4) * 16 + (c0 & 15), e = k0 & 15;
      *(uint2*)&ldsB[jt][l0 + 0][e] = q0;
      *(uint2*)&ldsB[jt][l0 + 1][e] = q1;
      *(uint2*)&ldsB[jt][l0 + 2][e] = q2;
      *(uint2*)&ldsB[jt][l0 + 3][e] = q3;
    }
    __syncthreads();
#pragma unroll
    for (int s = 0; s < 2; ++s)
      acc[s] = wmma_bf16(ldfrag(&ldsAt[i0[s] >> 4][lane][0]),
                         ldfrag(&ldsB[j0[s] >> 4][lane][0]), acc[s]);
    __syncthreads();
  }

  float* dst = fmp + ((size_t)b * KSPLIT + sk) * (FDIM * FDIM);
  const int hf = lane >> 4, nc = lane & 15;
#pragma unroll
  for (int s = 0; s < 2; ++s)
#pragma unroll
    for (int r = 0; r < 8; ++r)
      dst[(size_t)(i0[s] + r + 8 * hf) * FDIM + j0[s] + nc] = acc[s][r];
}

// ---------------------------------------------------------------------------
// Kernel 3: reduce split-K partials + row softmax over j; emit bf16.
// grid(B), block 64 (one row per thread; tiny).
// ---------------------------------------------------------------------------
__global__ __launch_bounds__(64) void k_softmax(const float* __restrict__ fmp,
                                                unsigned short* __restrict__ fmb) {
  const int b = blockIdx.x;
  const int i = threadIdx.x;
  float acc[FDIM];
#pragma unroll 8
  for (int j = 0; j < FDIM; ++j) acc[j] = 0.0f;
  for (int p = 0; p < KSPLIT; ++p) {
    const float* row = fmp + ((size_t)b * KSPLIT + p) * (FDIM * FDIM) + (size_t)i * FDIM;
#pragma unroll 8
    for (int j = 0; j < FDIM; ++j) acc[j] += row[j];
  }
  float mx = -3.4e38f;
#pragma unroll 8
  for (int j = 0; j < FDIM; ++j) mx = fmaxf(mx, acc[j]);
  float s = 0.0f;
#pragma unroll 8
  for (int j = 0; j < FDIM; ++j) { acc[j] = __expf(acc[j] - mx); s += acc[j]; }
  const float inv = 1.0f / s;
  unsigned short* orow = fmb + (size_t)b * FDIM * FDIM + (size_t)i * FDIM;
#pragma unroll 8
  for (int j = 0; j < FDIM; ++j) orow[j] = f2bf(acc[j] * inv);
}

// ---------------------------------------------------------------------------
// Kernel 4: z[b,j,n] = sum_i g[b,n,i] * fm[b,i,j]  -> zbuf bf16 [B, 64, N]
// grid(N/128, B), block 256.
// ---------------------------------------------------------------------------
__global__ __launch_bounds__(256) void k_attn(const unsigned short* __restrict__ g,
                                              const unsigned short* __restrict__ fmb,
                                              unsigned short* __restrict__ zbuf) {
  const int b  = blockIdx.y;
  const int n0 = blockIdx.x * 128;
  const int tid = threadIdx.x;
  const int lane = tid & 31, wave = tid >> 5;

  __shared__ __align__(32) unsigned short ldsA[8][2][32][16];  // g tile 128x64
  __shared__ __align__(32) unsigned short ldsB[2][4][32][16];  // fm 64x64

  // stage g tile: direct frag order (already bf16; pure dword copies)
#pragma unroll
  for (int it = 0; it < 8; ++it) {
    const int lin = tid * 4 + it * 1024;
    const int r = lin >> 6, c = lin & 63;
    const uint2 q = *(const uint2*)(g + ((size_t)b * NPIX + n0 + r) * FDIM + c);
    const int ks = c >> 5, kl = c & 31;
    *(uint2*)&ldsA[r >> 4][ks][(kl >> 4) * 16 + (r & 15)][kl & 15] = q;
  }
  // stage fm (64x64): one 4x4 perm-transpose block per thread
  {
    const int k0 = (tid >> 4) * 4, c0 = (tid & 15) * 4;
    const unsigned short* src = fmb + (size_t)b * FDIM * FDIM + (size_t)k0 * FDIM + c0;
    const uint2 r0 = *(const uint2*)(src);
    const uint2 r1 = *(const uint2*)(src + FDIM);
    const uint2 r2 = *(const uint2*)(src + 2 * FDIM);
    const uint2 r3 = *(const uint2*)(src + 3 * FDIM);
    uint2 q0, q1, q2, q3;
    tr4x4(r0, r1, r2, r3, q0, q1, q2, q3);
    const int ks = k0 >> 5, kl = k0 & 31;
    const int jt = c0 >> 4, l0 = (kl >> 4) * 16 + (c0 & 15), e = kl & 15;
    *(uint2*)&ldsB[ks][jt][l0 + 0][e] = q0;
    *(uint2*)&ldsB[ks][jt][l0 + 1][e] = q1;
    *(uint2*)&ldsB[ks][jt][l0 + 2][e] = q2;
    *(uint2*)&ldsB[ks][jt][l0 + 3][e] = q3;
  }
  __syncthreads();

  v8f acc[4];
#pragma unroll
  for (int j = 0; j < 4; ++j)
#pragma unroll
    for (int r = 0; r < 8; ++r) acc[j][r] = 0.0f;

#pragma unroll
  for (int ks = 0; ks < 2; ++ks) {
    const v16bf a = ldfrag(&ldsA[wave][ks][lane][0]);
#pragma unroll
    for (int jt = 0; jt < 4; ++jt)
      acc[jt] = wmma_bf16(a, ldfrag(&ldsB[ks][jt][lane][0]), acc[jt]);
  }

  const int hf = lane >> 4, nc = lane & 15;
#pragma unroll
  for (int jt = 0; jt < 4; ++jt) {
    const int j = jt * 16 + nc;
#pragma unroll
    for (int r = 0; r < 8; ++r) {
      const int m = r + 8 * hf;
      zbuf[(size_t)b * FDIM * NPIX + (size_t)j * NPIX + n0 + wave * 16 + m] = f2bf(acc[jt][r]);
    }
  }
}

// ---------------------------------------------------------------------------
// Kernel 5: flat-reshape gather + output conv + BN (fp32 out [B,N,64]).
// Output row p reads CONTIGUOUS slice zbuf[b, p>>8, (p&255)*64 .. +64].
//   sc = gamma * rsqrt(var+eps),  sh = (bo - mean)*sc + beta.
// ---------------------------------------------------------------------------
__global__ __launch_bounds__(256) void k_out(const unsigned short* __restrict__ zbuf,
                                             const float* __restrict__ Wo,
                                             const float* __restrict__ bo,
                                             const float* __restrict__ gamma,
                                             const float* __restrict__ beta,
                                             const float* __restrict__ mmean,
                                             const float* __restrict__ mvar,
                                             float* __restrict__ out) {
  const int b  = blockIdx.y;
  const int p0 = blockIdx.x * 128;
  const int tid = threadIdx.x;
  const int lane = tid & 31, wave = tid >> 5;

  __shared__ __align__(32) unsigned short ldsA[8][2][32][16];
  __shared__ __align__(32) unsigned short ldsB[2][4][32][16];
  __shared__ float sc[64], sh[64];

  // stage reshaped-z tile rows (contiguous gather), direct frag order
#pragma unroll
  for (int it = 0; it < 8; ++it) {
    const int lin = tid * 4 + it * 1024;
    const int r = lin >> 6, c = lin & 63;
    const int p = p0 + r;
    const unsigned short* src =
        zbuf + (size_t)b * FDIM * NPIX + (size_t)(p >> 8) * NPIX + (size_t)(p & 255) * FDIM + c;
    const uint2 q = *(const uint2*)src;
    const int ks = c >> 5, kl = c & 31;
    *(uint2*)&ldsA[r >> 4][ks][(kl >> 4) * 16 + (r & 15)][kl & 15] = q;
  }
  // stage Wo (64x64 fp32): pack rows, 4x4 perm transpose, one block per thread
  {
    const int k0 = (tid >> 4) * 4, c0 = (tid & 15) * 4;
    const float* ws = Wo + (size_t)k0 * FDIM + c0;
    const float4 w0 = *(const float4*)(ws);
    const float4 w1 = *(const float4*)(ws + FDIM);
    const float4 w2 = *(const float4*)(ws + 2 * FDIM);
    const float4 w3 = *(const float4*)(ws + 3 * FDIM);
    const uint2 a = {f2bf2(w0.x, w0.y), f2bf2(w0.z, w0.w)};
    const uint2 bb = {f2bf2(w1.x, w1.y), f2bf2(w1.z, w1.w)};
    const uint2 cc = {f2bf2(w2.x, w2.y), f2bf2(w2.z, w2.w)};
    const uint2 dd = {f2bf2(w3.x, w3.y), f2bf2(w3.z, w3.w)};
    uint2 q0, q1, q2, q3;
    tr4x4(a, bb, cc, dd, q0, q1, q2, q3);
    const int ks = k0 >> 5, kl = k0 & 31;
    const int jt = c0 >> 4, l0 = (kl >> 4) * 16 + (c0 & 15), e = kl & 15;
    *(uint2*)&ldsB[ks][jt][l0 + 0][e] = q0;
    *(uint2*)&ldsB[ks][jt][l0 + 1][e] = q1;
    *(uint2*)&ldsB[ks][jt][l0 + 2][e] = q2;
    *(uint2*)&ldsB[ks][jt][l0 + 3][e] = q3;
  }
  if (tid < 64) {
    const float s = gamma[tid] * rsqrtf(mvar[tid] + BN_EPS);
    sc[tid] = s;
    sh[tid] = (bo[tid] - mmean[tid]) * s + beta[tid];
  }
  __syncthreads();

  v8f acc[4];
#pragma unroll
  for (int j = 0; j < 4; ++j)
#pragma unroll
    for (int r = 0; r < 8; ++r) acc[j][r] = 0.0f;

#pragma unroll
  for (int ks = 0; ks < 2; ++ks) {
    const v16bf a = ldfrag(&ldsA[wave][ks][lane][0]);
#pragma unroll
    for (int jt = 0; jt < 4; ++jt)
      acc[jt] = wmma_bf16(a, ldfrag(&ldsB[ks][jt][lane][0]), acc[jt]);
  }

  const int hf = lane >> 4, nc = lane & 15;
#pragma unroll
  for (int jt = 0; jt < 4; ++jt) {
    const int col = jt * 16 + nc;
    const float s = sc[col], h = sh[col];
#pragma unroll
    for (int r = 0; r < 8; ++r) {
      const int m = r + 8 * hf;
      out[((size_t)b * NPIX + p0 + wave * 16 + m) * FDIM + col] = acc[jt][r] * s + h;
    }
  }
}

// ---------------------------------------------------------------------------
extern "C" void kernel_launch(void* const* d_in, const int* in_sizes, int n_in,
                              void* d_out, int out_size, void* d_ws, size_t ws_size,
                              hipStream_t stream) {
  const float* x     = (const float*)d_in[0];
  const float* Wg    = (const float*)d_in[1];
  const float* bg    = (const float*)d_in[2];
  const float* Wt    = (const float*)d_in[3];
  const float* bt    = (const float*)d_in[4];
  const float* Wp    = (const float*)d_in[5];
  const float* bp    = (const float*)d_in[6];
  const float* Wo    = (const float*)d_in[7];
  const float* bo    = (const float*)d_in[8];
  const float* gam   = (const float*)d_in[9];
  const float* bet   = (const float*)d_in[10];
  const float* mmean = (const float*)d_in[11];
  const float* mvar  = (const float*)d_in[12];
  float* out = (float*)d_out;

  // Workspace (elements): g_bf | tx_bf | ty_bf (bf16, 32 MiB each),
  //   fm_part fp32 [B][KSPLIT][64*64] (8 MiB), fmb bf16 [B][64*64].
  // zbuf aliases tx_bf (theta dead after k_gram).
  const size_t nbf = (size_t)BATCH * NPIX * FDIM;
  unsigned short* g_bf  = (unsigned short*)d_ws;
  unsigned short* tx_bf = g_bf + nbf;
  unsigned short* ty_bf = tx_bf + nbf;
  float*          fmp   = (float*)(ty_bf + nbf);
  unsigned short* fmb   = (unsigned short*)(fmp + (size_t)BATCH * KSPLIT * FDIM * FDIM);
  unsigned short* zbuf  = tx_bf;

  const dim3 blk(256);
  const dim3 grdP(NPIX / 128, BATCH);

  k_proj<<<grdP, blk, 0, stream>>>(x, Wg, bg, g_bf);
  k_proj<<<grdP, blk, 0, stream>>>(x, Wt, bt, tx_bf);
  k_proj<<<grdP, blk, 0, stream>>>(x, Wp, bp, ty_bf);
  k_gram<<<dim3(KSPLIT, BATCH), blk, 0, stream>>>(tx_bf, ty_bf, fmp);
  k_softmax<<<dim3(BATCH), dim3(64), 0, stream>>>(fmp, fmb);
  k_attn<<<grdP, blk, 0, stream>>>(g_bf, fmb, zbuf);
  k_out<<<grdP, blk, 0, stream>>>(zbuf, Wo, bo, gam, bet, mmean, mvar, out);
}
